// Attention_30674656428367
// MI455X (gfx1250) — matrix operand order repaired and checked by reading, not compiled, not run on hardware
//
#include <hip/hip_runtime.h>
#include <hip/hip_bf16.h>

#define NUM_HEADS 16
#define HEAD_DIM  64
#define SEQ_N     2048
#define BATCH     4
#define DMODEL    (NUM_HEADS * HEAD_DIM)

#define BM 64        // queries per workgroup (4 waves x 16 rows)
#define BN 32        // keys per inner iteration
#define KSTRIDE 88   // halves per K-tile row in LDS (16B-aligned rows, conflict-free b128)
#define VSTRIDE 40   // halves per V^T row in LDS (16B-aligned rows, conflict-free b128)
#define KTILE (32 * KSTRIDE)
#define VTILE (64 * VSTRIDE)

typedef __attribute__((ext_vector_type(16))) __bf16   v16bf;
typedef __attribute__((ext_vector_type(8)))  float    v8f;
typedef __attribute__((ext_vector_type(4)))  unsigned u32x4;
typedef __attribute__((ext_vector_type(4)))  float    f32x4;

union H16 { u32x4 u4[2]; unsigned u[8]; v16bf v; __bf16 h[16]; };
union F8  { v8f v; float f[8]; };
union PK2 { unsigned u; __bf16 h[2]; };

// raw v_exp_f32 (2^x), no libm denorm fixup — fine for softmax weights
static __device__ __forceinline__ float exp2_raw(float x) {
#if __has_builtin(__builtin_amdgcn_exp2f)
    return __builtin_amdgcn_exp2f(x);
#else
    return __builtin_exp2f(x);
#endif
}

// lane <-> lane^16 swap (cross-half), pure VALU on gfx1250
static __device__ __forceinline__ unsigned lane_xor16_u32(unsigned x) {
#if __has_builtin(__builtin_amdgcn_permlanex16)
    return (unsigned)__builtin_amdgcn_permlanex16((int)x, (int)x,
                      (int)0x76543210, (int)0xFEDCBA98, false, false);
#else
    return (unsigned)__shfl_xor((int)x, 16);
#endif
}
static __device__ __forceinline__ float lane_xor16(float x) {
    return __builtin_bit_cast(float, lane_xor16_u32(__builtin_bit_cast(unsigned, x)));
}
// lane <-> lane^4 swap (within 16-lane half), pure VALU
static __device__ __forceinline__ float lane_xor4(float x) {
#if __has_builtin(__builtin_amdgcn_permlane16)
    int i = __builtin_bit_cast(int, x);
    i = __builtin_amdgcn_permlane16(i, i, (int)0x32107654, (int)0xBA98FEDC, false, false);
    return __builtin_bit_cast(float, i);
#else
    return __shfl_xor(x, 4);
#endif
}

// Cooperative tile load: K row-major fp32->bf16, V transposed (packed b32 stores).
static __device__ __forceinline__ void load_tile(
    const float* __restrict__ kSrc, const float* __restrict__ vSrc,
    __bf16* __restrict__ sK, __bf16* __restrict__ sVT,
    const int ldKey, const int ldD, const bool evenSel, const bool prefetchNext)
{
    const f32x4* k4 = (const f32x4*)kSrc;
    const f32x4* v4 = (const f32x4*)vSrc;
    float kf[16], vf[16];
    #pragma unroll
    for (int j = 0; j < 4; ++j) {
        f32x4 a = k4[j];
        kf[j*4]=a.x; kf[j*4+1]=a.y; kf[j*4+2]=a.z; kf[j*4+3]=a.w;
        f32x4 c = v4[j];
        vf[j*4]=c.x; vf[j*4+1]=c.y; vf[j*4+2]=c.z; vf[j*4+3]=c.w;
    }
    // K: two vectorized 16B stores
    H16 kt16;
    #pragma unroll
    for (int i = 0; i < 16; ++i) kt16.h[i] = (__bf16)kf[i];
    u32x4* dst = (u32x4*)&sK[ldKey * KSTRIDE + ldD];
    dst[0] = kt16.u4[0]; dst[1] = kt16.u4[1];

    // V^T: pair lanes (t, t^4) hold keys (2j, 2j+1) with identical ldD.
    // Exchange the half of vf the partner needs, pack 2 keys/dword, 8 b32 stores.
    const int kp = ldKey & ~1;
    #pragma unroll
    for (int i = 0; i < 8; ++i) {
        const float snd = evenSel ? vf[8 + i] : vf[i];      // what partner needs
        const float x   = lane_xor4(snd);                    // what we receive
        const int   row = ldD + (evenSel ? i : 8 + i);       // d-row this lane stores
        PK2 p;
        p.h[0] = (__bf16)(evenSel ? vf[i] : x);              // even key value
        p.h[1] = (__bf16)(evenSel ? x : vf[8 + i]);          // odd key value
        *(unsigned*)&sVT[row * VSTRIDE + kp] = p.u;
    }
    if (prefetchNext) {
        __builtin_prefetch(kSrc + (size_t)BN * DMODEL, 0, 1);
        __builtin_prefetch(vSrc + (size_t)BN * DMODEL, 0, 1);
    }
}

// One 32-key flash-attention step. Scale is pre-folded into Q (log2 domain).
template<bool MASKED>
static __device__ __forceinline__ void fa_step(
    const int kv0, const int qg, const int hs, const int ln,
    const __bf16* __restrict__ sK, const __bf16* __restrict__ sVT,
    const H16 (&qf)[2], F8 (&o)[4], float& m, float& l)
{
    // ---- S^T = K * Q^T : two 16-key subtiles, head-dim as two K=32 chunks ----
    F8 acc[2];
    #pragma unroll
    for (int kk = 0; kk < 2; ++kk) {
        H16 ka0, ka1;
        const int row = kk * 16 + ln;
        ka0.u4[0] = *(const u32x4*)&sK[row * KSTRIDE +      hs * 8];
        ka0.u4[1] = *(const u32x4*)&sK[row * KSTRIDE + 16 + hs * 8];
        ka1.u4[0] = *(const u32x4*)&sK[row * KSTRIDE + 32 + hs * 8];
        ka1.u4[1] = *(const u32x4*)&sK[row * KSTRIDE + 48 + hs * 8];
        F8 a;
        #pragma unroll
        for (int r = 0; r < 8; ++r) a.f[r] = 0.0f;
        a.v = __builtin_amdgcn_wmma_f32_16x16x32_bf16(false, ka0.v, false, qf[0].v,
                                                      (short)0, a.v, false, false);
        a.v = __builtin_amdgcn_wmma_f32_16x16x32_bf16(false, ka1.v, false, qf[1].v,
                                                      (short)0, a.v, false, false);
        acc[kk] = a;
    }

    // ---- causal mask in place (boundary tiles only) ----
    if (MASKED) {
        #pragma unroll
        for (int r = 0; r < 8; ++r) {
            const int key0 = kv0 + hs * 8 + r;
            acc[0].f[r] = (key0      <= qg) ? acc[0].f[r] : -1e30f;
            acc[1].f[r] = (key0 + 16 <= qg) ? acc[1].f[r] : -1e30f;
        }
    }

    // ---- online softmax in log2 domain ----
    float mt = -1e30f;
    #pragma unroll
    for (int r = 0; r < 8; ++r)
        mt = fmaxf(mt, fmaxf(acc[0].f[r], acc[1].f[r]));
    mt = fmaxf(mt, lane_xor16(mt));
    const float mnew  = fmaxf(m, mt);
    const float alpha = exp2_raw(m - mnew);
    float p0[8], p1[8], psum = 0.0f;
    #pragma unroll
    for (int r = 0; r < 8; ++r) {
        p0[r] = exp2_raw(acc[0].f[r] - mnew);
        p1[r] = exp2_raw(acc[1].f[r] - mnew);
        psum += p0[r] + p1[r];
    }
    psum += lane_xor16(psum);
    l = l * alpha + psum;
    m = mnew;

    // ---- build P^T B-fragment: pack to bf16 dwords first, then one permlane
    //      and one select per dword (keys 2j,2j+1 live in dword j) ----
    H16 pf;
    #pragma unroll
    for (int j = 0; j < 4; ++j) {
        PK2 t0, t1;
        t0.h[0] = (__bf16)p0[2*j]; t0.h[1] = (__bf16)p0[2*j+1];
        t1.h[0] = (__bf16)p1[2*j]; t1.h[1] = (__bf16)p1[2*j+1];
        const unsigned x0 = lane_xor16_u32(t0.u);
        const unsigned x1 = lane_xor16_u32(t1.u);
        pf.u[j]     = hs ? x1   : t0.u;   // keys 0..7   | 16..23
        pf.u[4 + j] = hs ? t1.u : x0;     // keys 8..15  | 24..31
    }

    // ---- O^T = alpha*O^T + V^T * P^T (4 output d-tiles) ----
    #pragma unroll
    for (int t = 0; t < 4; ++t) {
        H16 va;
        const int d = t * 16 + ln;
        va.u4[0] = *(const u32x4*)&sVT[d * VSTRIDE +     hs * 16];
        va.u4[1] = *(const u32x4*)&sVT[d * VSTRIDE + 8 + hs * 16];
        #pragma unroll
        for (int r = 0; r < 8; ++r) o[t].f[r] *= alpha;
        o[t].v = __builtin_amdgcn_wmma_f32_16x16x32_bf16(false, va.v, false, pf.v,
                                                         (short)0, o[t].v, false, false);
    }
}

__global__ __launch_bounds__(128) void
fa_fwd_causal_kernel(const float* __restrict__ Q,
                     const float* __restrict__ K,
                     const float* __restrict__ V,
                     float* __restrict__ O)
{
    // double-buffered tiles
    __shared__ __align__(16) __bf16 sK [2 * KTILE];   // K tiles, row-major [key][d]
    __shared__ __align__(16) __bf16 sVT[2 * VTILE];   // V tiles, transposed [d][key]

    const int tid  = threadIdx.x;
    const int lane = tid & 31;
    const int hs   = lane >> 4;
    const int ln   = lane & 15;
    const int wave = tid >> 5;

    const int qblk = blockIdx.x & 31;        // N / BM = 32 blocks per head
    const int bh   = blockIdx.x >> 5;
    const int h    = bh & (NUM_HEADS - 1);
    const int b    = bh >> 4;

    const int q0 = qblk * BM;
    const int qg = q0 + wave * 16 + ln;      // this lane's query (WMMA column N)

    const size_t headoff = (size_t)b * SEQ_N * DMODEL + (size_t)h * HEAD_DIM;

    // ---- Q B-fragments, pre-scaled by 1/sqrt(64)*log2(e) (log2-domain softmax) ----
    const float qScale = 0.125f * 1.44269504088896f;
    H16 qf[2];
    {
        const float* qrow = Q + headoff + (size_t)qg * DMODEL;
        #pragma unroll
        for (int c = 0; c < 2; ++c) {
            const f32x4* s4 = (const f32x4*)(qrow + c * 32 + hs * 8);
            #pragma unroll
            for (int j = 0; j < 4; ++j) {
                f32x4 f = s4[j + (j & 2)];
                qf[c].h[j * 4 + 0] = (__bf16)(f.x * qScale);
                qf[c].h[j * 4 + 1] = (__bf16)(f.y * qScale);
                qf[c].h[j * 4 + 2] = (__bf16)(f.z * qScale);
                qf[c].h[j * 4 + 3] = (__bf16)(f.w * qScale);
            }
        }
    }

    // cooperative K/V tile-load mapping: 128 threads cover 32 keys x 64 dims
    const int  ldKey   = tid >> 2;            // 0..31
    const int  ldD     = (tid & 3) * 16;      // 0,16,32,48
    const bool evenSel = ((lane >> 2) & 1) == 0;

    F8 o[4];
    #pragma unroll
    for (int t = 0; t < 4; ++t)
        #pragma unroll
        for (int r = 0; r < 8; ++r) o[t].f[r] = 0.0f;
    float m = -1e30f, l = 0.0f;

    const int maskStart = 2 * qblk;           // first key-tile crossing the diagonal
    const int nkt       = maskStart + 2;
    const float* kBase = K + headoff + (size_t)ldKey * DMODEL + ldD;
    const float* vBase = V + headoff + (size_t)ldKey * DMODEL + ldD;

    // preload tile 0 into buffer 0
    load_tile(kBase, vBase, sK, sVT, ldKey, ldD, evenSel, true);

    // ---- steady state: fully-unmasked key tiles (double-buffered, 1 barrier/iter) ----
    for (int kt = 0; kt < maskStart; ++kt) {
        __syncthreads();
        const int nb = (kt + 1) & 1;
        load_tile(kBase + (size_t)(kt + 1) * BN * DMODEL,
                  vBase + (size_t)(kt + 1) * BN * DMODEL,
                  sK + nb * KTILE, sVT + nb * VTILE,
                  ldKey, ldD, evenSel, kt + 2 < nkt);
        const int cb = kt & 1;
        fa_step<false>(kt * BN, qg, hs, ln, sK + cb * KTILE, sVT + cb * VTILE, qf, o, m, l);
    }
    // ---- last two tiles: diagonal, causal-masked ----
    #pragma unroll
    for (int kk = 0; kk < 2; ++kk) {
        const int kt = maskStart + kk;
        __syncthreads();
        if (kk == 0) {
            const int nb = (kt + 1) & 1;
            load_tile(kBase + (size_t)(kt + 1) * BN * DMODEL,
                      vBase + (size_t)(kt + 1) * BN * DMODEL,
                      sK + nb * KTILE, sVT + nb * VTILE,
                      ldKey, ldD, evenSel, false);
        }
        const int cb = kt & 1;
        fa_step<true>(kt * BN, qg, hs, ln, sK + cb * KTILE, sVT + cb * VTILE, qf, o, m, l);
    }

    // ---- epilogue: normalize and store (lane qg owns d = t*16 + hs*8 + r) ----
    const float inv = 1.0f / l;
    float* obase = O + headoff + (size_t)qg * DMODEL;
    #pragma unroll
    for (int t = 0; t < 4; ++t) {
        float* dp = obase + t * 16 + hs * 8;
        f32x4 lo, hi;
        lo.x = o[t].f[0] * inv; lo.y = o[t].f[1] * inv;
        lo.z = o[t].f[2] * inv; lo.w = o[t].f[3] * inv;
        hi.x = o[t].f[4] * inv; hi.y = o[t].f[5] * inv;
        hi.z = o[t].f[6] * inv; hi.w = o[t].f[7] * inv;
        *(f32x4*)dp       = lo;
        *(f32x4*)(dp + 4) = hi;
    }
}

extern "C" void kernel_launch(void* const* d_in, const int* in_sizes, int n_in,
                              void* d_out, int out_size, void* d_ws, size_t ws_size,
                              hipStream_t stream) {
    const float* q = (const float*)d_in[0];
    const float* k = (const float*)d_in[1];
    const float* v = (const float*)d_in[2];
    // d_in[3] = causal mask (bool) — causality handled analytically in-kernel.
    float* out = (float*)d_out;
    const int blocks = BATCH * NUM_HEADS * (SEQ_N / BM);  // 2048
    fa_fwd_causal_kernel<<<dim3(blocks), dim3(128), 0, stream>>>(q, k, v, out);
}
